// EdgeAwareMHSA_60782377173530
// MI455X (gfx1250) — compile-verified, matrix operand order checked by reading, not run on hardware
//
#include <hip/hip_runtime.h>
#include <hip/hip_bf16.h>
#include <math.h>

typedef __attribute__((ext_vector_type(16))) __bf16 v16bf;
typedef __attribute__((ext_vector_type(8)))  float  v8f;

#define S_LEN  2048
#define BATCH  2
#define DMODEL 1024
#define NHEADS 16
#define HD     64
#define MROWS  4096   // S*B

union FragU {
  v16bf v;
  unsigned int u[8];
};

// fp32 -> bf16, round-to-nearest-even
__device__ __forceinline__ unsigned short f2bf(float f) {
  unsigned int u = __builtin_bit_cast(unsigned int, f);
  u += 0x7FFFu + ((u >> 16) & 1u);
  return (unsigned short)(u >> 16);
}

// Load a 16x32 bf16 tile (row-major, strideH bf16 elements between rows, strideH even)
// into the wave32 WMMA A/B fragment layout:
//   lane L holds row M = L%16; lanes 0-15 get K-blocks {0..7,16..23}, lanes 16-31 {8..15,24..31}.
// Per lane this is two contiguous 16B reads -> compiler emits global_load_b128 pairs.
__device__ __forceinline__ v16bf load_frag16(const unsigned short* base, int strideH) {
  const int lane = threadIdx.x & 31;
  const int m    = lane & 15;
  const int koff = (lane & 16) ? 8 : 0;
  const unsigned int* p32 = (const unsigned int*)base;
  const int rowOff = (m * strideH) >> 1;           // dwords
  FragU f;
#pragma unroll
  for (int j = 0; j < 8; ++j) {
    const int k = ((j < 4) ? (2 * j) : (16 + 2 * (j - 4))) + koff;
    f.u[j] = p32[rowOff + (k >> 1)];
  }
  return f.v;
}

// ---------------------------------------------------------------------------
// One-time fp32 -> bf16 copy (src, ctx stays bf16 already)
// ---------------------------------------------------------------------------
__global__ __launch_bounds__(256)
void cvt_copy_kernel(const float* __restrict__ in, unsigned short* __restrict__ out, int n) {
  const int i = blockIdx.x * 256 + threadIdx.x;
  if (i < n) out[i] = f2bf(in[i]);
}

// ---------------------------------------------------------------------------
// One-time fp32 -> bf16 transpose for weights: out[n*K + k] = bf16(in[k*N + n])
// Classic 32x32 LDS tile transpose; block (32,8).
// ---------------------------------------------------------------------------
__global__ __launch_bounds__(256)
void cvt_transpose_kernel(const float* __restrict__ in, unsigned short* __restrict__ out) {
  __shared__ unsigned short tile[32][33];
  const int tx = threadIdx.x, ty = threadIdx.y;
  const int k0 = blockIdx.x * 32;   // input row block
  const int n0 = blockIdx.y * 32;   // input col block
#pragma unroll
  for (int j = 0; j < 4; ++j) {
    const int k = ty + j * 8;
    tile[k][tx] = f2bf(in[(size_t)(k0 + k) * DMODEL + n0 + tx]);
  }
  __syncthreads();
#pragma unroll
  for (int j = 0; j < 4; ++j) {
    const int n = ty + j * 8;
    out[(size_t)(n0 + n) * DMODEL + k0 + tx] = tile[tx][n];
  }
}

// ---------------------------------------------------------------------------
// GEMM: C[M=4096, N=1024] = A[4096,1024](bf16) @ Wt[1024,1024](bf16, [n][k]) + bias
// No LDS: fragments loaded straight from global (L2-resident bf16 operands).
// Block 256 threads = 8 waves; wave tile 32x64; block tile 256x64; K-step 32.
// MODE 0: out -> Q  bf16 [b,h,s,hd]
// MODE 1: out -> K  bf16 [b,h,s,hd]
// MODE 2: out -> Vt bf16 [b,h,hd,s]   (transposed for attention)
// MODE 3: out -> fp32 [s,b,d] (d_out)
// ---------------------------------------------------------------------------
template <int MODE>
__global__ __launch_bounds__(256)
void gemm_bf16_kernel(const unsigned short* __restrict__ A,
                      const unsigned short* __restrict__ Wt,
                      const float* __restrict__ bias,
                      unsigned short* __restrict__ outH,
                      float* __restrict__ outF)
{
  const int tid  = threadIdx.x;
  const int wave = tid >> 5;
  const int lane = tid & 31;
  const int row0 = blockIdx.x * 256 + wave * 32;
  const int nn   = blockIdx.y * 64;

  v8f acc[2][4] = {};

  for (int kk = 0; kk < DMODEL; kk += 32) {
    const v16bf a0 = load_frag16(A + (size_t)row0 * DMODEL + kk, DMODEL);
    const v16bf a1 = load_frag16(A + (size_t)(row0 + 16) * DMODEL + kk, DMODEL);
#pragma unroll
    for (int nt = 0; nt < 4; ++nt) {
      const v16bf bfr = load_frag16(Wt + (size_t)(nn + nt * 16) * DMODEL + kk, DMODEL);
      acc[0][nt] = __builtin_amdgcn_wmma_f32_16x16x32_bf16(
          false, a0, false, bfr, (short)0, acc[0][nt], false, false);
      acc[1][nt] = __builtin_amdgcn_wmma_f32_16x16x32_bf16(
          false, a1, false, bfr, (short)0, acc[1][nt], false, false);
    }
  }

  const int half = lane >> 4;
  const int nloc = lane & 15;
#pragma unroll
  for (int rt = 0; rt < 2; ++rt) {
#pragma unroll
    for (int nt = 0; nt < 4; ++nt) {
      const int col = nn + nt * 16 + nloc;
      const float bv = bias[col];
#pragma unroll
      for (int r = 0; r < 8; ++r) {
        const int row = row0 + rt * 16 + r + 8 * half;   // row = s*B + b
        const float val = acc[rt][nt][r] + bv;
        if (MODE == 3) {
          outF[(size_t)row * DMODEL + col] = val;
        } else {
          const int b = row & 1, s = row >> 1;
          const int h = col >> 6, e = col & 63;
          if (MODE == 2)
            outH[((size_t)(b * NHEADS + h) * HD + e) * S_LEN + s] = f2bf(val);
          else
            outH[((size_t)(b * NHEADS + h) * S_LEN + s) * HD + e] = f2bf(val);
        }
      }
    }
  }
}

// ---------------------------------------------------------------------------
// Flash attention: one wave per 16-row q-tile per (b,h). 32 keys / iteration.
// logits = (Q Kt)*0.125 + bias ; online softmax ; p_signed = p*sign(bias)
// out = (sum p_signed V) / (sum p)
// ---------------------------------------------------------------------------
__global__ __launch_bounds__(32)
void attn_kernel(const unsigned short* __restrict__ qb,
                 const unsigned short* __restrict__ kb,
                 const unsigned short* __restrict__ vtb,
                 const float* __restrict__ eb,
                 unsigned short* __restrict__ ctx)
{
  __shared__ unsigned short pls[16][40];   // signed-p re-swizzle buffer
  const int lane = threadIdx.x & 31;
  const int half = lane >> 4;
  const int nloc = lane & 15;
  const int s0 = blockIdx.x * 16;
  const int bh = blockIdx.y;               // b*16 + h

  const unsigned short* qrow = qb + ((size_t)bh * S_LEN + s0) * HD;
  const v16bf qlo = load_frag16(qrow, HD);       // d = 0..31
  const v16bf qhi = load_frag16(qrow + 32, HD);  // d = 32..63

  float runm[8], runl[8];
#pragma unroll
  for (int r = 0; r < 8; ++r) { runm[r] = -INFINITY; runl[r] = 0.f; }
  v8f acc[4] = {};

  for (int kk = 0; kk < S_LEN; kk += 32) {
    v8f lg[2];
#pragma unroll
    for (int j = 0; j < 2; ++j) {
      const unsigned short* krow = kb + ((size_t)bh * S_LEN + kk + 16 * j) * HD;
      const v16bf klo = load_frag16(krow, HD);
      const v16bf khi = load_frag16(krow + 32, HD);
      v8f c = {};
      c = __builtin_amdgcn_wmma_f32_16x16x32_bf16(false, qlo, false, klo, (short)0, c, false, false);
      c = __builtin_amdgcn_wmma_f32_16x16x32_bf16(false, qhi, false, khi, (short)0, c, false, false);
      lg[j] = c;
    }
    // scale + edge bias + sign capture (element (m=r+8*half, t=kk+16j+nloc))
    float sgn[2][8];
#pragma unroll
    for (int j = 0; j < 2; ++j) {
#pragma unroll
      for (int r = 0; r < 8; ++r) {
        const float bv = eb[(size_t)(s0 + r + 8 * half) * S_LEN + kk + 16 * j + nloc];
        lg[j][r] = lg[j][r] * 0.125f + bv;
        sgn[j][r] = (bv < 0.f) ? -1.f : 1.f;
      }
    }
    // online softmax: rows live in 16-lane halves -> xor-reduce over {8,4,2,1}
#pragma unroll
    for (int r = 0; r < 8; ++r) {
      float tm = fmaxf(lg[0][r], lg[1][r]);
#pragma unroll
      for (int off = 8; off >= 1; off >>= 1)
        tm = fmaxf(tm, __shfl_xor(tm, off, 32));
      const float nm   = fmaxf(runm[r], tm);
      const float corr = __expf(runm[r] - nm);
      const float p0 = __expf(lg[0][r] - nm);
      const float p1 = __expf(lg[1][r] - nm);
      float rs = p0 + p1;
#pragma unroll
      for (int off = 8; off >= 1; off >>= 1)
        rs += __shfl_xor(rs, off, 32);
      runl[r] = runl[r] * corr + rs;
      runm[r] = nm;
#pragma unroll
      for (int nt = 0; nt < 4; ++nt) acc[nt][r] *= corr;
      pls[r + 8 * half][nloc]      = f2bf(p0 * sgn[0][r]);
      pls[r + 8 * half][16 + nloc] = f2bf(p1 * sgn[1][r]);
    }
    __syncthreads();
    const v16bf pA = load_frag16(&pls[0][0], 40);   // 16 rows x 32 keys
#pragma unroll
    for (int nt = 0; nt < 4; ++nt) {
      const unsigned short* vrow = vtb + ((size_t)bh * HD + nt * 16) * S_LEN + kk;
      const v16bf vB = load_frag16(vrow, S_LEN);    // lane=d col, K along s
      acc[nt] = __builtin_amdgcn_wmma_f32_16x16x32_bf16(
          false, pA, false, vB, (short)0, acc[nt], false, false);
    }
    __syncthreads();
  }

  const int h = bh & 15, b = bh >> 4;
#pragma unroll
  for (int r = 0; r < 8; ++r) {
    const float inv = 1.f / runl[r];
    const int s = s0 + r + 8 * half;
#pragma unroll
    for (int nt = 0; nt < 4; ++nt) {
      const int d = nt * 16 + nloc;
      ctx[((size_t)s * BATCH + b) * DMODEL + h * HD + d] = f2bf(acc[nt][r] * inv);
    }
  }
}

extern "C" void kernel_launch(void* const* d_in, const int* in_sizes, int n_in,
                              void* d_out, int out_size, void* d_ws, size_t ws_size,
                              hipStream_t stream) {
  const float* src   = (const float*)d_in[0];
  const float* ebias = (const float*)d_in[1];
  const float* wq    = (const float*)d_in[2];
  const float* bq    = (const float*)d_in[3];
  const float* wk    = (const float*)d_in[4];
  const float* bk    = (const float*)d_in[5];
  const float* wv    = (const float*)d_in[6];
  const float* bv    = (const float*)d_in[7];
  const float* wo    = (const float*)d_in[8];
  const float* bo    = (const float*)d_in[9];
  float* out = (float*)d_out;

  // Workspace (bf16 ushort elements):
  //   qb, kb, vtb, ctx : 4 x MROWS*DMODEL   (32 MB)
  //   srcB             : MROWS*DMODEL       ( 8 MB)
  //   wqT,wkT,wvT,woT  : 4 x DMODEL*DMODEL  ( 8 MB)
  unsigned short* qb   = (unsigned short*)d_ws;
  unsigned short* kb   = qb   + (size_t)MROWS * DMODEL;
  unsigned short* vtb  = kb   + (size_t)MROWS * DMODEL;
  unsigned short* ctx  = vtb  + (size_t)MROWS * DMODEL;
  unsigned short* srcB = ctx  + (size_t)MROWS * DMODEL;
  unsigned short* wqT  = srcB + (size_t)MROWS * DMODEL;
  unsigned short* wkT  = wqT  + (size_t)DMODEL * DMODEL;
  unsigned short* wvT  = wkT  + (size_t)DMODEL * DMODEL;
  unsigned short* woT  = wvT  + (size_t)DMODEL * DMODEL;

  // One-time conversions
  const int nSrc = MROWS * DMODEL;
  cvt_copy_kernel<<<(nSrc + 255) / 256, 256, 0, stream>>>(src, srcB, nSrc);
  dim3 tg(DMODEL / 32, DMODEL / 32), tb(32, 8);
  cvt_transpose_kernel<<<tg, tb, 0, stream>>>(wq, wqT);
  cvt_transpose_kernel<<<tg, tb, 0, stream>>>(wk, wkT);
  cvt_transpose_kernel<<<tg, tb, 0, stream>>>(wv, wvT);
  cvt_transpose_kernel<<<tg, tb, 0, stream>>>(wo, woT);

  dim3 gg(MROWS / 256, DMODEL / 64);
  gemm_bf16_kernel<0><<<gg, 256, 0, stream>>>(srcB, wqT, bq, qb,  nullptr);
  gemm_bf16_kernel<1><<<gg, 256, 0, stream>>>(srcB, wkT, bk, kb,  nullptr);
  gemm_bf16_kernel<2><<<gg, 256, 0, stream>>>(srcB, wvT, bv, vtb, nullptr);
  attn_kernel<<<dim3(S_LEN / 16, BATCH * NHEADS), 32, 0, stream>>>(qb, kb, vtb, ebias, ctx);
  gemm_bf16_kernel<3><<<gg, 256, 0, stream>>>(ctx, woT, bo, nullptr, out);
}
